// MDDGCN_82764019793947
// MI455X (gfx1250) — compile-verified, matrix operand order checked by reading, not run on hardware
//
#include <hip/hip_runtime.h>
#include <hip/hip_bf16.h>

// ---------------------------------------------------------------------------
// MDDGCN forward for MI455X (gfx1250, wave32, WMMA).
// deg -> dis -> attention scale -> 3x { edge-aggregate + WMMA GEMM (fused BN
// stats) + BN apply } + 2 dense skip branches + blends.
// B operands pre-packed into CDNA5 WMMA fragment layout (v16h per lane).
// ---------------------------------------------------------------------------

#define NNODES 50000
#define NEDGES 800000
#define BN_EPS 1e-5f

typedef __attribute__((ext_vector_type(16))) _Float16 v16h;
typedef __attribute__((ext_vector_type(8)))  float    v8f;

union V16H { v16h v; _Float16 h[16]; unsigned u[8]; };
union V8F  { v8f  v; float f[8]; };

// ---------------- utility kernels ----------------

__global__ void k_zero(float* __restrict__ p, long long n) {
  long long i = (long long)blockIdx.x * blockDim.x + threadIdx.x;
  for (; i < n; i += (long long)gridDim.x * blockDim.x) p[i] = 0.f;
}

// Pack one K=128 x C weight matrix (f32, row-major) into WMMA B-fragment
// layout: o[((ct*4 + ks)*32 + lane)*16 + j] — one v16h per (wave,kstep,lane).
__global__ void k_pack(const float* __restrict__ W, _Float16* __restrict__ o, int C) {
  int i = blockIdx.x * blockDim.x + threadIdx.x;
  int total = 128 * C;
  if (i >= total) return;
  int j    = i & 15;
  int lane = (i >> 4) & 31;
  int ks   = (i >> 9) & 3;
  int ct   = i >> 11;
  int col  = ct * 16 + (lane & 15);
  int base = (lane >> 4) ? 8 : 0;
  int p    = j >> 1;
  int k    = ks * 32 + ((p < 4) ? (base + 2 * p) : (base + 16 + 2 * (p - 4))) + (j & 1);
  o[i] = (_Float16)W[k * C + col];
}

__global__ void k_deg(const long long* __restrict__ ei, float* __restrict__ deg) {
  long long e = (long long)blockIdx.x * blockDim.x + threadIdx.x;
  if (e < NEDGES) atomicAdd(&deg[ei[e]], 1.0f);
}

__global__ void k_dis(const float* __restrict__ deg, float* __restrict__ dis) {
  int i = blockIdx.x * blockDim.x + threadIdx.x;
  if (i < NNODES) {
    float d = deg[i];
    dis[i] = (d > 0.f) ? rsqrtf(fmaxf(d, 1.0f)) : 0.f;
  }
}

__global__ void k_colsum(const float* __restrict__ x, float* __restrict__ colsum) {
  int col = threadIdx.x;  // blockDim.x == 128
  float s = 0.f;
  for (int r = blockIdx.x; r < NNODES; r += gridDim.x)
    s += x[(size_t)r * 128 + col];
  atomicAdd(&colsum[col], s);
}

__global__ void k_att(const float* __restrict__ colsum,
                      const float* __restrict__ w1, const float* __restrict__ b1,
                      const float* __restrict__ w2, const float* __restrict__ b2,
                      float* __restrict__ wfull) {
  __shared__ float mean[128];
  __shared__ float a1[128];
  __shared__ float wts[4];
  int t = threadIdx.x;
  mean[t] = colsum[t] * (1.0f / (float)NNODES);
  __syncthreads();
  float s = b1[t];
  for (int i = 0; i < 128; ++i) s += mean[i] * w1[i * 128 + t];
  a1[t] = s > 0.f ? s : 0.f;
  __syncthreads();
  if (t < 4) {
    float r = b2[t];
    for (int j = 0; j < 128; ++j) r += a1[j] * w2[j * 4 + t];
    wts[t] = r;
  }
  __syncthreads();
  if (t == 0) {
    float mx = fmaxf(fmaxf(wts[0], wts[1]), fmaxf(wts[2], wts[3]));
    float e0 = __expf(wts[0] - mx), e1 = __expf(wts[1] - mx);
    float e2 = __expf(wts[2] - mx), e3 = __expf(wts[3] - mx);
    float inv = 1.f / (e0 + e1 + e2 + e3);
    wts[0] = e0 * inv; wts[1] = e1 * inv; wts[2] = e2 * inv; wts[3] = e3 * inv;
  }
  __syncthreads();
  wfull[t] = wts[t >> 5];   // FEATURE_DIMS = (32,32,32,32)
}

__global__ void k_scale(const float* __restrict__ x, const float* __restrict__ wfull,
                        float* __restrict__ h) {
  long long i = (long long)blockIdx.x * blockDim.x + threadIdx.x;
  if (i < (long long)NNODES * 128) h[i] = x[i] * wfull[i & 127];
}

// edge aggregation: one wave per edge, 4 floats per lane (float4 gather)
__global__ void k_agg(const long long* __restrict__ ei, const float* __restrict__ dis,
                      const float* __restrict__ h, float* __restrict__ tx) {
  long long t = (long long)blockIdx.x * blockDim.x + threadIdx.x;
  long long e = t >> 5;
  int lane = (int)(t & 31);
  if (e >= NEDGES) return;
  long long s = ei[e];
  long long d = ei[NEDGES + e];
  float w = -dis[s] * dis[d];
  const float4 v = *(const float4*)&h[s * 128 + lane * 4];
  float* o = &tx[d * 128 + lane * 4];
  atomicAdd(o + 0, w * v.x);
  atomicAdd(o + 1, w * v.y);
  atomicAdd(o + 2, w * v.z);
  atomicAdd(o + 3, w * v.w);
}

// ---------------- WMMA GEMM: out = A@B0 (+ A2@B1) + bias, fused BN stats ----
// blockDim.x = 32*(C/16); grid.x = NNODES/16. A,A2: N x 128 f32.
// B0,B1: pre-packed fragment layout (k_pack).
template <int USEA2, int DORELU>
__global__ __launch_bounds__(256) void k_gemm(
    const float* __restrict__ A, const float* __restrict__ A2,
    const _Float16* __restrict__ B0, const _Float16* __restrict__ B1,
    const float* __restrict__ bias, float* __restrict__ out,
    float* __restrict__ ssum, float* __restrict__ ssq, int C)
{
  __shared__ _Float16 As[16][128];
  __shared__ _Float16 Ts[16][128];
  const int tid = threadIdx.x;
  const int nthreads = blockDim.x;
  const int row0 = blockIdx.x * 16;

  for (int i = tid; i < 16 * 128; i += nthreads) {
    int r = i >> 7, c = i & 127;
    As[r][c] = (_Float16)A[(size_t)(row0 + r) * 128 + c];
    if (USEA2) Ts[r][c] = (_Float16)A2[(size_t)(row0 + r) * 128 + c];
  }
  __syncthreads();

  const int wave = tid >> 5;
  const int lane = tid & 31;
  const int c0   = wave * 16;
  const int col  = c0 + (lane & 15);
  const int hi   = lane >> 4;
  const int arow = lane & 15;

  V8F acc;
  #pragma unroll
  for (int i = 0; i < 8; ++i) acc.f[i] = 0.f;

  // A fragment K-offsets: contiguous dword pairs -> ds_load_b128 x2 per frag
  const int kbaseA = hi ? 8 : 0;

  #pragma unroll
  for (int ks = 0; ks < 4; ++ks) {
    V16H a, t;
    #pragma unroll
    for (int p = 0; p < 4; ++p) {
      int kA = ks * 32 + kbaseA + 2 * p;
      a.u[p]     = *(const unsigned*)&As[arow][kA];
      a.u[p + 4] = *(const unsigned*)&As[arow][kA + 16];
      if (USEA2) {
        t.u[p]     = *(const unsigned*)&Ts[arow][kA];
        t.u[p + 4] = *(const unsigned*)&Ts[arow][kA + 16];
      }
    }
    const size_t fbase = (((size_t)(wave * 4 + ks) * 32) + lane) << 4;
    const v16h b0 = *(const v16h*)(B0 + fbase);
    acc.v = __builtin_amdgcn_wmma_f32_16x16x32_f16(false, a.v, false, b0,
                                                   (short)0, acc.v, false, false);
    if (USEA2) {
      const v16h b1 = *(const v16h*)(B1 + fbase);
      acc.v = __builtin_amdgcn_wmma_f32_16x16x32_f16(false, t.v, false, b1,
                                                     (short)0, acc.v, false, false);
    }
  }

  const float bcol = bias[col];
  float lsum = 0.f, lsq = 0.f;
  #pragma unroll
  for (int r = 0; r < 8; ++r) {
    float v = acc.f[r] + bcol;
    if (DORELU) v = v > 0.f ? v : 0.f;
    int m = r + (hi ? 8 : 0);
    out[(size_t)(row0 + m) * C + col] = v;
    lsum += v; lsq += v * v;
  }
  float osum = __shfl_xor(lsum, 16, 32);
  float osq  = __shfl_xor(lsq, 16, 32);
  if (lane < 16) {
    atomicAdd(&ssum[col], lsum + osum);
    atomicAdd(&ssq[col], lsq + osq);
  }
}

__global__ void k_bnfinal(const float* __restrict__ ssum, const float* __restrict__ ssq,
                          const float* __restrict__ g, const float* __restrict__ b,
                          float* __restrict__ scale, float* __restrict__ shift, int C) {
  int c = blockIdx.x * blockDim.x + threadIdx.x;
  if (c < C) {
    float m = ssum[c] * (1.0f / (float)NNODES);
    float v = ssq[c] * (1.0f / (float)NNODES) - m * m;
    float sc = g[c] * rsqrtf(v + BN_EPS);
    scale[c] = sc;
    shift[c] = b[c] - m * sc;
  }
}

__global__ void k_bn_relu(const float* __restrict__ z, const float* __restrict__ scale,
                          const float* __restrict__ shift, float* __restrict__ o, int C) {
  long long i = (long long)blockIdx.x * blockDim.x + threadIdx.x;
  if (i < (long long)NNODES * C) {
    int c = (int)(i % C);
    float v = z[i] * scale[c] + shift[c];
    o[i] = v > 0.f ? v : 0.f;
  }
}

// h = relu((1-am)*relu(bn2(z2)) + am*bnm(rm))
__global__ void k_blend(const float* __restrict__ z2, const float* __restrict__ rm,
                        const float* __restrict__ s2, const float* __restrict__ sh2,
                        const float* __restrict__ sm, const float* __restrict__ shm,
                        const float* __restrict__ amp, float* __restrict__ o) {
  long long i = (long long)blockIdx.x * blockDim.x + threadIdx.x;
  if (i < (long long)NNODES * 128) {
    int c = (int)(i & 127);
    float am = *amp;
    float h2 = z2[i] * s2[c] + sh2[c];
    h2 = h2 > 0.f ? h2 : 0.f;
    float mv = rm[i] * sm[c] + shm[c];
    float v = (1.f - am) * h2 + am * mv;
    o[i] = v > 0.f ? v : 0.f;
  }
}

// out = relu((1-aa)*bn3(z3) + aa*bna(ra)), C = 64
__global__ void k_final(const float* __restrict__ z3, const float* __restrict__ ra,
                        const float* __restrict__ s3, const float* __restrict__ sh3,
                        const float* __restrict__ sa, const float* __restrict__ sha,
                        const float* __restrict__ aap, float* __restrict__ o) {
  long long i = (long long)blockIdx.x * blockDim.x + threadIdx.x;
  if (i < (long long)NNODES * 64) {
    int c = (int)(i & 63);
    float aa = *aap;
    float h3 = z3[i] * s3[c] + sh3[c];
    float av = ra[i] * sa[c] + sha[c];
    float v = (1.f - aa) * h3 + aa * av;
    o[i] = v > 0.f ? v : 0.f;
  }
}

// ---------------------------------------------------------------------------

extern "C" void kernel_launch(void* const* d_in, const int* in_sizes, int n_in,
                              void* d_out, int out_size, void* d_ws, size_t ws_size,
                              hipStream_t stream) {
  const float*     x    = (const float*)d_in[0];
  const long long* ei   = (const long long*)d_in[1];
  const float* att_w1 = (const float*)d_in[2];
  const float* att_b1 = (const float*)d_in[3];
  const float* att_w2 = (const float*)d_in[4];
  const float* att_b2 = (const float*)d_in[5];
  const float* c1_w   = (const float*)d_in[6];
  const float* c1_b   = (const float*)d_in[7];
  const float* bn1_g  = (const float*)d_in[8];
  const float* bn1_b  = (const float*)d_in[9];
  const float* c2_w   = (const float*)d_in[10];
  const float* c2_b   = (const float*)d_in[11];
  const float* bn2_g  = (const float*)d_in[12];
  const float* bn2_b  = (const float*)d_in[13];
  const float* c3_w   = (const float*)d_in[14];
  const float* c3_b   = (const float*)d_in[15];
  const float* bn3_g  = (const float*)d_in[16];
  const float* bn3_b  = (const float*)d_in[17];
  const float* fcm_w  = (const float*)d_in[18];
  const float* fcm_b  = (const float*)d_in[19];
  const float* bnm_g  = (const float*)d_in[20];
  const float* bnm_b  = (const float*)d_in[21];
  const float* fca_w  = (const float*)d_in[22];
  const float* fca_b  = (const float*)d_in[23];
  const float* bna_g  = (const float*)d_in[24];
  const float* bna_b  = (const float*)d_in[25];
  const float* am_p   = (const float*)d_in[26];
  const float* aa_p   = (const float*)d_in[27];
  float* out = (float*)d_out;

  float* ws = (float*)d_ws;
  const long long NF = (long long)NNODES * 128;
  float* deg   = ws;
  float* dis   = ws + NNODES;
  float* small = ws + 2 * NNODES;
  float* colsum = small + 0;
  float* wfull  = small + 128;
  float* sumA   = small + 256;   // +sqA contiguous
  float* sqA    = small + 384;
  float* scaleA = small + 512;
  float* shiftA = small + 640;
  float* sumB   = small + 768;   // +sqB contiguous
  float* sqB    = small + 896;
  float* scaleB = small + 1024;
  float* shiftB = small + 1152;
  float* BUF0 = small + 2048;          // h current
  float* BUF1 = BUF0 + NF;             // tx1
  float* BUF2 = BUF1 + NF;             // gemm out
  float* BUF3 = BUF2 + NF;             // skip branch
  _Float16* hw = (_Float16*)(BUF3 + NF);
  _Float16* c1h  = hw;                 // 2*128*128
  _Float16* c2h  = hw + 32768;         // 2*128*128
  _Float16* fcmh = hw + 65536;         // 128*128
  _Float16* c3h  = hw + 81920;         // 2*128*64
  _Float16* fcah = hw + 98304;         // 128*64

  const int T = 256;
  dim3 b256(T);
  int nTileBlocks = NNODES / 16;       // 3125
  long long aggThreads = (long long)NEDGES * 32;
  int aggBlocks = (int)((aggThreads + T - 1) / T);
  int nfBlocks = (int)((NF + T - 1) / T);
  int n64Blocks = (int)(((long long)NNODES * 64 + T - 1) / T);

  // 0) zero deg + small region
  k_zero<<<1024, b256, 0, stream>>>(deg, NNODES);
  k_zero<<<8, b256, 0, stream>>>(small, 2048);

  // 1) degree / dis
  k_deg<<<(NEDGES + T - 1) / T, b256, 0, stream>>>(ei, deg);
  k_dis<<<(NNODES + T - 1) / T, b256, 0, stream>>>(deg, dis);

  // 2) attention feature scale
  k_colsum<<<256, dim3(128), 0, stream>>>(x, colsum);
  k_att<<<1, dim3(128), 0, stream>>>(colsum, att_w1, att_b1, att_w2, att_b2, wfull);
  k_scale<<<nfBlocks, b256, 0, stream>>>(x, wfull, BUF0);

  // 3) pack weights into WMMA B-fragment layout (half)
  k_pack<<<64, b256, 0, stream>>>(c1_w,          c1h,          128);
  k_pack<<<64, b256, 0, stream>>>(c1_w + 16384,  c1h + 16384,  128);
  k_pack<<<64, b256, 0, stream>>>(c2_w,          c2h,          128);
  k_pack<<<64, b256, 0, stream>>>(c2_w + 16384,  c2h + 16384,  128);
  k_pack<<<64, b256, 0, stream>>>(fcm_w,         fcmh,         128);
  k_pack<<<32, b256, 0, stream>>>(c3_w,          c3h,          64);
  k_pack<<<32, b256, 0, stream>>>(c3_w + 8192,   c3h + 8192,   64);
  k_pack<<<32, b256, 0, stream>>>(fca_w,         fcah,         64);

  // ---- cheb layer 1 ----
  k_zero<<<4096, b256, 0, stream>>>(BUF1, NF);
  k_agg<<<aggBlocks, b256, 0, stream>>>(ei, dis, BUF0, BUF1);
  k_zero<<<1, b256, 0, stream>>>(sumA, 256);
  k_gemm<1, 0><<<nTileBlocks, b256, 0, stream>>>(BUF0, BUF1, c1h, c1h + 16384, c1_b,
                                                 BUF2, sumA, sqA, 128);
  k_bnfinal<<<1, dim3(128), 0, stream>>>(sumA, sqA, bn1_g, bn1_b, scaleA, shiftA, 128);
  k_bn_relu<<<nfBlocks, b256, 0, stream>>>(BUF2, scaleA, shiftA, BUF0, 128);  // h1

  // ---- cheb layer 2 ----
  k_zero<<<4096, b256, 0, stream>>>(BUF1, NF);
  k_agg<<<aggBlocks, b256, 0, stream>>>(ei, dis, BUF0, BUF1);
  k_zero<<<1, b256, 0, stream>>>(sumA, 256);
  k_gemm<1, 0><<<nTileBlocks, b256, 0, stream>>>(BUF0, BUF1, c2h, c2h + 16384, c2_b,
                                                 BUF2, sumA, sqA, 128);
  k_bnfinal<<<1, dim3(128), 0, stream>>>(sumA, sqA, bn2_g, bn2_b, scaleA, shiftA, 128);

  // ---- main skip branch: rm = relu(x@fcm + b), stats over rm ----
  k_zero<<<1, b256, 0, stream>>>(sumB, 256);
  k_gemm<0, 1><<<nTileBlocks, b256, 0, stream>>>(x, x, fcmh, fcmh, fcm_b,
                                                 BUF3, sumB, sqB, 128);
  k_bnfinal<<<1, dim3(128), 0, stream>>>(sumB, sqB, bnm_g, bnm_b, scaleB, shiftB, 128);

  // ---- blend -> h3 ----
  k_blend<<<nfBlocks, b256, 0, stream>>>(BUF2, BUF3, scaleA, shiftA, scaleB, shiftB,
                                         am_p, BUF0);

  // ---- cheb layer 3 (C = 64) ----
  k_zero<<<4096, b256, 0, stream>>>(BUF1, NF);
  k_agg<<<aggBlocks, b256, 0, stream>>>(ei, dis, BUF0, BUF1);
  k_zero<<<1, b256, 0, stream>>>(sumA, 256);
  k_gemm<1, 0><<<nTileBlocks, dim3(128), 0, stream>>>(BUF0, BUF1, c3h, c3h + 8192, c3_b,
                                                      BUF2, sumA, sqA, 64);
  k_bnfinal<<<1, dim3(64), 0, stream>>>(sumA, sqA, bn3_g, bn3_b, scaleA, shiftA, 64);

  // ---- aux skip branch: ra = relu(x@fca + b) (C = 64) ----
  k_zero<<<1, b256, 0, stream>>>(sumB, 256);
  k_gemm<0, 1><<<nTileBlocks, dim3(128), 0, stream>>>(x, x, fcah, fcah, fca_b,
                                                      BUF3, sumB, sqB, 64);
  k_bnfinal<<<1, dim3(64), 0, stream>>>(sumB, sqB, bna_g, bna_b, scaleB, shiftB, 64);

  // ---- final blend -> d_out ----
  k_final<<<n64Blocks, b256, 0, stream>>>(BUF2, BUF3, scaleA, shiftA, scaleB, shiftB,
                                          aa_p, out);
}
